// RNN2_20126216749784
// MI455X (gfx1250) — matrix operand order repaired and checked by reading, not compiled, not run on hardware
//
#include <hip/hip_runtime.h>

typedef __bf16 bf16_t;
typedef __bf16 v16bf __attribute__((ext_vector_type(16)));
typedef float  v8f   __attribute__((ext_vector_type(8)));

union V16 {
  v16bf v;
  uint4 q[2];
  bf16_t e[16];
};

#define B_  256
#define S_  512
#define E_  100
#define H_  128
#define KP  128     // K padded to 128
#define G3  384     // 3*H

// ---------------- kernel 0: f32 [rows,ksrc] -> bf16 [rows,kdst] with zero pad
__global__ void cvt_pad_bf16(const float* __restrict__ src, bf16_t* __restrict__ dst,
                             int rows, int ksrc, int kdst) {
  int idx = blockIdx.x * blockDim.x + threadIdx.x;
  int total = rows * kdst;
  if (idx >= total) return;
  int r = idx / kdst;
  int k = idx - r * kdst;
  float v = (k < ksrc) ? src[r * ksrc + k] : 0.0f;
  dst[idx] = (bf16_t)v;
}

// ---------------- kernel 1: XG[t, 0:384] = emb[tok[t]] @ W_ih^T + b_ih  (bf16 out)
// grid: (B*S)/16 workgroups of 128 threads (4 waves). Wave w owns N in [96w, 96w+96).
// All A and B tiles are hoisted into registers BEFORE the WMMA chain so the
// 56 b128 loads pipeline as clauses instead of one wait per WMMA.
__global__ void __launch_bounds__(128)
xg_gemm(const int* __restrict__ tok, const bf16_t* __restrict__ embb,
        const bf16_t* __restrict__ wih, const float* __restrict__ b_ih,
        bf16_t* __restrict__ xg) {
  const int lane = threadIdx.x & 31;
  const int wave = threadIdx.x >> 5;     // 0..3
  const int m    = lane & 15;
  const int hi   = lane >> 4;            // 0/1
  const int t0   = blockIdx.x * 16;

  // token whose embedding row this lane supplies as A-matrix row m
  const int myTok = tok[t0 + m];

  // A-matrix (16x32 bf16) for 4 K-steps: two 16B runs per lane per K-step
  V16 A[4];
#pragma unroll
  for (int ks = 0; ks < 4; ++ks) {
    const bf16_t* p = embb + (size_t)myTok * KP + ks * 32 + hi * 8;
    A[ks].q[0] = *(const uint4*)p;
    A[ks].q[1] = *(const uint4*)(p + 16);
  }

  // B tiles (W_ih^T), all hoisted: 6 N-tiles x 4 K-steps
  V16 Bt[6][4];
#pragma unroll
  for (int nt = 0; nt < 6; ++nt) {
    const int n0 = wave * 96 + nt * 16;
    const bf16_t* row = wih + (size_t)(n0 + m) * KP + hi * 8;
#pragma unroll
    for (int ks = 0; ks < 4; ++ks) {
      Bt[nt][ks].q[0] = *(const uint4*)(row + ks * 32);
      Bt[nt][ks].q[1] = *(const uint4*)(row + ks * 32 + 16);
    }
  }

  v8f acc[6];
#pragma unroll
  for (int nt = 0; nt < 6; ++nt) {
    v8f c = {};
#pragma unroll
    for (int ks = 0; ks < 4; ++ks)
      c = __builtin_amdgcn_wmma_f32_16x16x32_bf16(false, A[ks].v, false, Bt[nt][ks].v,
                                                  (short)0, c, false, false);
    acc[nt] = c;
  }

  // add b_ih and store bf16 XG (C layout: VGPR v, lane l -> M=v+8*hi, N=l&15)
#pragma unroll
  for (int nt = 0; nt < 6; ++nt) {
    const int n = wave * 96 + nt * 16 + m;
    const float bias = b_ih[n];
#pragma unroll
    for (int v = 0; v < 8; ++v) {
      const int row = t0 + v + 8 * hi;
      xg[(size_t)row * G3 + n] = (bf16_t)(acc[nt][v] + bias);
    }
  }
}

// ---------------- kernel 2: GRU scan ----------------------------------------
// Each workgroup owns 16 batch rows (fully independent -> no global sync).
// 8 waves; wave w owns hidden columns [16w, 16w+16). W_hh lives in B-layout
// registers; h lives in C-layout registers + bf16 LDS mirror for the A side.
// XG loads are software-pipelined one timestep ahead in registers so the
// serial per-step chain is only: barrier -> ds_load A -> WMMA -> VALU -> ds_store.

__device__ __forceinline__ void load_xg_regs(const bf16_t* __restrict__ xg, int s,
                                             int b0, int hi, int j, float (&x)[3][8]) {
#pragma unroll
  for (int v = 0; v < 8; ++v) {
    const int bm = v + 8 * hi;
    const size_t base = ((size_t)(b0 + bm) * S_ + s) * G3;
    x[0][v] = (float)xg[base + j];
    x[1][v] = (float)xg[base + H_ + j];
    x[2][v] = (float)xg[base + 2 * H_ + j];
  }
}

__device__ __forceinline__ void gru_step(int s, const bf16_t* __restrict__ xg,
                                         bf16_t* hbuf, const V16 (&W)[3][4],
                                         float (&h)[8], float (&xcur)[3][8],
                                         float (&xnxt)[3][8],
                                         int b0, int m, int hi, int j,
                                         float bh_r, float bh_z, float bh_n) {
  // A = current h (16x128 bf16) from LDS, A-matrix layout
  V16 A[4];
#pragma unroll
  for (int ks = 0; ks < 4; ++ks) {
    const bf16_t* p = &hbuf[m * KP + ks * 32 + hi * 8];
    A[ks].q[0] = *(const uint4*)p;
    A[ks].q[1] = *(const uint4*)(p + 16);
  }
  __syncthreads();  // all waves consumed hbuf before it is overwritten

  v8f ar = {}, az = {}, an = {};
#pragma unroll
  for (int ks = 0; ks < 4; ++ks) {
    ar = __builtin_amdgcn_wmma_f32_16x16x32_bf16(false, A[ks].v, false, W[0][ks].v,
                                                 (short)0, ar, false, false);
    az = __builtin_amdgcn_wmma_f32_16x16x32_bf16(false, A[ks].v, false, W[1][ks].v,
                                                 (short)0, az, false, false);
    an = __builtin_amdgcn_wmma_f32_16x16x32_bf16(false, A[ks].v, false, W[2][ks].v,
                                                 (short)0, an, false, false);
  }

  // prefetch next step's gate pre-activations into registers (consumed next call)
  const int sp = (s + 1 < S_) ? s + 1 : s;
  load_xg_regs(xg, sp, b0, hi, j, xnxt);

  // gate math + h update using the values prefetched LAST step
#pragma unroll
  for (int v = 0; v < 8; ++v) {
    const int bm = v + 8 * hi;
    const float r  = 1.0f / (1.0f + __expf(-(xcur[0][v] + ar[v] + bh_r)));
    const float z  = 1.0f / (1.0f + __expf(-(xcur[1][v] + az[v] + bh_z)));
    const float ng = tanhf(xcur[2][v] + r * (an[v] + bh_n));
    h[v] = (1.0f - z) * ng + z * h[v];
    hbuf[bm * KP + j] = (bf16_t)h[v];
  }
  __syncthreads();
}

__global__ void __launch_bounds__(256)
gru_scan(const bf16_t* __restrict__ xg, const bf16_t* __restrict__ whh,
         const float* __restrict__ h0, const float* __restrict__ b_hh,
         const float* __restrict__ dec_w, const float* __restrict__ dec_b,
         float* __restrict__ out) {
  __shared__ __align__(16) bf16_t hbuf[16 * KP];  // current h tile, bf16, 4 KB
  __shared__ float outred[16];

  const int lane = threadIdx.x & 31;
  const int wave = threadIdx.x >> 5;     // 0..7
  const int m    = lane & 15;
  const int hi   = lane >> 4;
  const int b0   = blockIdx.x * 16;
  const int j    = wave * 16 + m;        // hidden column owned by this lane slot

  // W_hh resident in registers, B-matrix layout: gate g, K-step ks
  V16 W[3][4];
#pragma unroll
  for (int g = 0; g < 3; ++g)
#pragma unroll
    for (int ks = 0; ks < 4; ++ks) {
      const bf16_t* p = whh + (size_t)(g * H_ + j) * KP + ks * 32 + hi * 8;
      W[g][ks].q[0] = *(const uint4*)p;
      W[g][ks].q[1] = *(const uint4*)(p + 16);
    }

  const float bh_r = b_hh[j];
  const float bh_z = b_hh[H_ + j];
  const float bh_n = b_hh[2 * H_ + j];

  // init h in C layout registers + bf16 LDS mirror
  float h[8];
#pragma unroll
  for (int v = 0; v < 8; ++v) {
    const int bm = v + 8 * hi;
    h[v] = h0[(size_t)(b0 + bm) * H_ + j];
    hbuf[bm * KP + j] = (bf16_t)h[v];
  }
  __syncthreads();

  // software pipeline: xa/xb ping-pong, time loop unrolled by 2
  float xa[3][8], xb[3][8];
  load_xg_regs(xg, 0, b0, hi, j, xa);

  for (int s = 0; s < S_; s += 2) {
    gru_step(s,     xg, hbuf, W, h, xa, xb, b0, m, hi, j, bh_r, bh_z, bh_n);
    gru_step(s + 1, xg, hbuf, W, h, xb, xa, b0, m, hi, j, bh_r, bh_z, bh_n);
  }

  // decoder head: out[b] = h[b,:] . dec_w + dec_b
  if (threadIdx.x < 16) outred[threadIdx.x] = 0.0f;
  __syncthreads();
  const float dw = dec_w[j];
#pragma unroll
  for (int v = 0; v < 8; ++v) atomicAdd(&outred[v + 8 * hi], h[v] * dw);
  __syncthreads();
  if (threadIdx.x < 16) out[b0 + threadIdx.x] = outred[threadIdx.x] + dec_b[0];
}

extern "C" void kernel_launch(void* const* d_in, const int* in_sizes, int n_in,
                              void* d_out, int out_size, void* d_ws, size_t ws_size,
                              hipStream_t stream) {
  const int*   inputs = (const int*)d_in[0];
  const float* h0     = (const float*)d_in[1];
  const float* emb    = (const float*)d_in[2];
  const float* W_ih   = (const float*)d_in[3];
  const float* W_hh   = (const float*)d_in[4];
  const float* b_ih   = (const float*)d_in[5];
  const float* b_hh   = (const float*)d_in[6];
  const float* dec_w  = (const float*)d_in[7];
  const float* dec_b  = (const float*)d_in[8];
  float* out = (float*)d_out;

  // workspace layout (bytes, all 256-aligned):
  //   emb_bf16 : 50000*128*2 = 12,800,000
  //   wih_bf16 :   384*128*2 =     98,304
  //   whh_bf16 :   384*128*2 =     98,304
  //   xg_bf16  : 131072*384*2 = 100,663,296   (total ~113.7 MB)
  char* ws = (char*)d_ws;
  bf16_t* embb = (bf16_t*)(ws);
  bf16_t* wihb = (bf16_t*)(ws + 12800000);
  bf16_t* whhb = (bf16_t*)(ws + 12898304);
  bf16_t* xgp  = (bf16_t*)(ws + 12996608);

  const int n_emb = 50000 * KP;
  cvt_pad_bf16<<<(n_emb + 255) / 256, 256, 0, stream>>>(emb, embb, 50000, E_, KP);
  cvt_pad_bf16<<<(G3 * KP + 255) / 256, 256, 0, stream>>>(W_ih, wihb, G3, E_, KP);
  cvt_pad_bf16<<<(G3 * KP + 255) / 256, 256, 0, stream>>>(W_hh, whhb, G3, H_, KP);

  xg_gemm<<<(B_ * S_) / 16, 128, 0, stream>>>(inputs, embb, wihb, b_ih, xgp);
  gru_scan<<<B_ / 16, 256, 0, stream>>>(xgp, whhb, h0, b_hh, dec_w, dec_b, out);
}
